// CrAKNEncoder_49813030699158
// MI455X (gfx1250) — compile-verified, matrix operand order checked by reading, not run on hardware
//
#include <hip/hip_runtime.h>
#include <hip/hip_bf16.h>
#include <math.h>

// ---------------------------------------------------------------------------
// Problem constants (from reference)
// ---------------------------------------------------------------------------
#define H_   4
#define HD_  64
#define BD_  40
#define QDIM_ 256
#define NQ_  1024
#define NK_  1024
#define D_   256        // H*HD
#define HBD_ 160        // H*BD
#define NPAIR_ (NQ_*NK_)   // 1048576

typedef __bf16 bf16_t;
typedef __attribute__((ext_vector_type(16))) __bf16 v16bf;
typedef __attribute__((ext_vector_type(8)))  float  v8f;

// ---------------------------------------------------------------------------
// WMMA helpers (gfx1250: D = A(16x32 bf16) x B(32x16 bf16) + C(16x16 f32))
// ---------------------------------------------------------------------------
__device__ __forceinline__ v8f wmma_bf16(v16bf a, v16bf b, v8f c) {
  return __builtin_amdgcn_wmma_f32_16x16x32_bf16(
      /*neg_a=*/false, a, /*neg_b=*/false, b,
      /*c_mod=*/(short)0, c, /*reuse_a=*/false, /*reuse_b=*/false);
}

// A operand: row-major [M x K] bf16, tile at (row0, k0).  Lane l: m = l&15,
// half = l>>4.  Elements 0..7 -> K = k0 + half*8 + e ; 8..15 -> K = k0+16+half*8+(e-8).
// B operand uses the same loader on a TRANSPOSED (N x K row-major) matrix.
__device__ __forceinline__ v16bf load_frag_bf16(const bf16_t* src, int ld,
                                                int row0, int k0, int lane) {
  int m = lane & 15, half = lane >> 4;
  const bf16_t* p = src + (long)(row0 + m) * ld + k0 + half * 8;
  v16bf a;
#pragma unroll
  for (int e = 0; e < 8; ++e) a[e] = p[e];
#pragma unroll
  for (int e = 0; e < 8; ++e) a[e + 8] = p[16 + e];
  return a;
}

// Fragment from fp32 source (LDS tile), converted on the fly.
__device__ __forceinline__ v16bf load_frag_f32cvt(const float* src, int ld,
                                                  int row0, int k0, int lane) {
  int m = lane & 15, half = lane >> 4;
  const float* p = src + (long)(row0 + m) * ld + k0 + half * 8;
  v16bf a;
#pragma unroll
  for (int e = 0; e < 8; ++e) a[e] = (bf16_t)p[e];
#pragma unroll
  for (int e = 0; e < 8; ++e) a[e + 8] = (bf16_t)p[16 + e];
  return a;
}

// Fragment from fp32 source with K zero-padding (kk >= kmax -> 0).
// Loads are index-clamped so no out-of-bounds LDS access is emitted.
__device__ __forceinline__ v16bf load_frag_f32cvt_pad(const float* src, int ld, int kmax,
                                                      int row0, int k0, int lane) {
  int m = lane & 15, half = lane >> 4;
  const float* p = src + (long)(row0 + m) * ld;
  v16bf a;
#pragma unroll
  for (int e = 0; e < 16; ++e) {
    int kk = k0 + half * 8 + ((e < 8) ? e : (e + 8));
    int kc = (kk < kmax) ? kk : 0;
    float v = p[kc];
    a[e] = (kk < kmax) ? (bf16_t)v : (bf16_t)0.f;
  }
  return a;
}

__device__ __forceinline__ float mish_f(float x) {
  float sp = (x > 20.f) ? x : log1pf(__expf(x));
  return x * tanhf(sp);
}

// ---------------------------------------------------------------------------
// Prep kernels
// ---------------------------------------------------------------------------
__global__ void k_cvt_bf16(const float* __restrict__ in, bf16_t* __restrict__ out, int n) {
  for (int i = blockIdx.x * blockDim.x + threadIdx.x; i < n; i += gridDim.x * blockDim.x)
    out[i] = (bf16_t)in[i];
}

// in: [K x N] fp32 (weights stored [in,out]) -> out: [Npad x Kpad] bf16 (B^T, zero pad)
__global__ void k_transpose_pad_bf16(const float* __restrict__ in, bf16_t* __restrict__ out,
                                     int K, int N, int Kpad, int Npad) {
  int tot = Npad * Kpad;
  for (int i = blockIdx.x * blockDim.x + threadIdx.x; i < tot; i += gridDim.x * blockDim.x) {
    int n = i / Kpad, kk = i - n * Kpad;
    float v = (n < N && kk < K) ? in[(long)kk * N + n] : 0.f;
    out[i] = (bf16_t)v;
  }
}

// Wqo/Wko: fold torch permute(1,2,0) flatten: pq[:, hd*H+h] == proj_flat[:, h*HD+hd]
// out[n*256 + (h*64+hd)] = in[(hd*4+h)*256 + n]   (transposed + row-permuted, bf16)
__global__ void k_perm_transpose(const float* __restrict__ in, bf16_t* __restrict__ out) {
  for (int i = blockIdx.x * blockDim.x + threadIdx.x; i < D_ * D_; i += gridDim.x * blockDim.x) {
    int n = i >> 8, c = i & 255;
    int h = c >> 6, hd = c & 63;
    out[i] = (bf16_t)in[(hd * H_ + h) * D_ + n];
  }
}

// rv[k] = sum_v v[k, :]
__global__ void k_rowsum(const float* __restrict__ v, float* __restrict__ rv) {
  int row = blockIdx.x * blockDim.x + threadIdx.x;
  if (row < NK_) {
    float s = 0.f;
    for (int c = 0; c < QDIM_; ++c) s += v[(long)row * QDIM_ + c];
    rv[row] = s;
  }
}

// ---------------------------------------------------------------------------
// Projection GEMM: out_bf = bf16(A @ B^T + bias), A:[1024x256]bf16, Bt:[256x256]bf16
// one 16x16 tile per wave; 8 waves per block; grid = 128
// ---------------------------------------------------------------------------
__global__ __launch_bounds__(256) void k_proj(const bf16_t* __restrict__ A,
                                              const bf16_t* __restrict__ Bt,
                                              const float* __restrict__ bvec,
                                              bf16_t* __restrict__ outbf) {
  int lane = threadIdx.x & 31, wave = threadIdx.x >> 5;
  int tile = blockIdx.x * 8 + wave;          // 1024 tiles = 64 (M) x 16 (N)
  int m0 = (tile >> 4) * 16, n0 = (tile & 15) * 16;
  v8f acc = {};
#pragma unroll
  for (int kt = 0; kt < 8; ++kt) {
    v16bf a = load_frag_bf16(A, D_, m0, kt * 32, lane);
    v16bf b = load_frag_bf16(Bt, D_, n0, kt * 32, lane);
    acc = wmma_bf16(a, b, acc);
  }
  int n = n0 + (lane & 15), half = lane >> 4;
  float bv = bvec[n];
#pragma unroll
  for (int r = 0; r < 8; ++r) {
    int row = m0 + r + half * 8;
    outbf[(long)row * D_ + n] = (bf16_t)(acc[r] + bv);
  }
}

// ---------------------------------------------------------------------------
// Fused bias path: per 16-pair tile t (wave-owned):
//   stage bias tile (2560 contiguous bytes) via GLOBAL_LOAD_ASYNC_TO_LDS_B128
//   Y = bias_tile[16x40] @ Wbias + bbias      (bf16 WMMA, K padded 40->64)
//   diffs[h, p] = ||Y[:, h*40:(h+1)*40]||_2
//   bias_out    = mish(Y @ Wbo + bbo)         (bf16 WMMA, K=160)
// 4 waves / 128 threads per block; grid = NPAIR/(4*16) = 16384
// ---------------------------------------------------------------------------
__global__ __launch_bounds__(128) void k_bias(const float* __restrict__ bias,
                                              const float* __restrict__ bbias,
                                              const float* __restrict__ bbo,
                                              const bf16_t* __restrict__ Wbias_t,  // [160 x 64]
                                              const bf16_t* __restrict__ Wbo_t,    // [48 x 160]
                                              float* __restrict__ diffs,           // [4 x 1048576]
                                              float* __restrict__ bias_out) {      // [1048576 x 40]
  __shared__ float sB[4][16 * 40];   // fp32 bias tiles (async-copied)
  __shared__ float sY[4][16 * 160];
  int lane = threadIdx.x & 31, wave = threadIdx.x >> 5;
  long p0 = ((long)blockIdx.x * 4 + wave) * 16;

  // --- async stage: 16 rows x 40 f32 = 2560 contiguous bytes -> LDS ---
  // Low 32 bits of a flat shared pointer are the LDS offset (aperture rule);
  // the inst offset applies to both the global and LDS sides.
  {
    unsigned ldsoff = (unsigned)(uintptr_t)(&sB[wave][0]) + lane * 16u;
    const char* gsrc = (const char*)(bias + p0 * BD_) + lane * 16;
    asm volatile(
        "global_load_async_to_lds_b128 %0, %1, off\n\t"
        "global_load_async_to_lds_b128 %0, %1, off offset:512\n\t"
        "global_load_async_to_lds_b128 %0, %1, off offset:1024\n\t"
        "global_load_async_to_lds_b128 %0, %1, off offset:1536\n\t"
        "global_load_async_to_lds_b128 %0, %1, off offset:2048\n\t"
        "s_wait_asynccnt 0"
        :: "v"(ldsoff), "v"(gsrc) : "memory");
  }
  __syncthreads();

  int nloc = lane & 15, half = lane >> 4;
  // A fragments from fp32 LDS tile (ld=40), K zero-padded 40 -> 64
  v16bf a0 = load_frag_f32cvt_pad(&sB[wave][0], BD_, BD_, 0, 0, lane);
  v16bf a1 = load_frag_f32cvt_pad(&sB[wave][0], BD_, BD_, 0, 32, lane);
#pragma unroll
  for (int nt = 0; nt < 10; ++nt) {
    v16bf b0 = load_frag_bf16(Wbias_t, 64, nt * 16, 0, lane);
    v16bf b1 = load_frag_bf16(Wbias_t, 64, nt * 16, 32, lane);
    v8f acc = {};
    acc = wmma_bf16(a0, b0, acc);
    acc = wmma_bf16(a1, b1, acc);
    int n = nt * 16 + nloc;
    float bb = bbias[n];
#pragma unroll
    for (int r = 0; r < 8; ++r) sY[wave][(r + half * 8) * HBD_ + n] = acc[r] + bb;
  }
  __syncthreads();

  // per-head L2 norms -> diffs
  for (int idx = lane; idx < 64; idx += 32) {
    int r = idx >> 2, h = idx & 3;
    float s = 0.f;
    for (int c = 0; c < BD_; ++c) {
      float t = sY[wave][r * HBD_ + h * BD_ + c];
      s += t * t;
    }
    diffs[(long)h * NPAIR_ + p0 + r] = sqrtf(s);
  }

  // Y @ Wbo (K=160, N=40 padded to 48), + bbo, mish, store
  v16bf ya[5];
#pragma unroll
  for (int kt = 0; kt < 5; ++kt)
    ya[kt] = load_frag_f32cvt(&sY[wave][0], HBD_, 0, kt * 32, lane);
#pragma unroll
  for (int nt = 0; nt < 3; ++nt) {
    v8f acc = {};
#pragma unroll
    for (int kt = 0; kt < 5; ++kt) {
      v16bf b = load_frag_bf16(Wbo_t, HBD_, nt * 16, kt * 32, lane);
      acc = wmma_bf16(ya[kt], b, acc);
    }
    int n = nt * 16 + nloc;
    if (n < BD_) {
      float bb = bbo[n];
#pragma unroll
      for (int r = 0; r < 8; ++r) {
        long p = p0 + r + half * 8;
        bias_out[p * BD_ + n] = mish_f(acc[r] + bb);
      }
    }
  }
}

// ---------------------------------------------------------------------------
// Attention: per (head, 16-query tile): logits via WMMA (+diffs, mask),
// softmax over 1024, then dot with rv[k] = rowsum(v)  ->  vals_mean partials.
// grid (64, 4), 256 threads (8 waves each own 128 key-columns of logits).
// ---------------------------------------------------------------------------
__global__ __launch_bounds__(256) void k_attn(const bf16_t* __restrict__ pq,
                                              const bf16_t* __restrict__ pk,
                                              const float* __restrict__ diffs,
                                              const int* __restrict__ mask,
                                              const float* __restrict__ rv,
                                              float* __restrict__ vm_accum) {
  __shared__ float sl[16 * 1024];   // 64 KB logits tile
  int h = blockIdx.y;
  int q0 = blockIdx.x * 16;
  int lane = threadIdx.x & 31, wave = threadIdx.x >> 5;
  int nloc = lane & 15, half = lane >> 4;

  v16bf a0 = load_frag_bf16(pq, D_, q0, h * HD_, lane);
  v16bf a1 = load_frag_bf16(pq, D_, q0, h * HD_ + 32, lane);
#pragma unroll
  for (int i = 0; i < 8; ++i) {
    int n0 = (wave * 8 + i) * 16;
    v16bf b0 = load_frag_bf16(pk, D_, n0, h * HD_, lane);
    v16bf b1 = load_frag_bf16(pk, D_, n0, h * HD_ + 32, lane);
    v8f acc = {};
    acc = wmma_bf16(a0, b0, acc);
    acc = wmma_bf16(a1, b1, acc);
    int n = n0 + nloc;
#pragma unroll
    for (int r = 0; r < 8; ++r) {
      int row = r + half * 8;
      int q = q0 + row;
      float lg = acc[r] * 0.125f + diffs[(long)h * NPAIR_ + (long)q * NK_ + n];
      if (mask[(long)q * NK_ + n] == 0) lg = -9e15f;
      sl[row * NK_ + n] = lg;
    }
  }
  __syncthreads();

  // softmax + dot(rv): 16 threads per query row
  int row = threadIdx.x >> 4, l16 = threadIdx.x & 15;
  float mx = -INFINITY;
  for (int c = l16; c < NK_; c += 16) mx = fmaxf(mx, sl[row * NK_ + c]);
#pragma unroll
  for (int o = 8; o; o >>= 1) mx = fmaxf(mx, __shfl_xor(mx, o, 16));
  float se = 0.f, dot = 0.f;
  for (int c = l16; c < NK_; c += 16) {
    float e = __expf(sl[row * NK_ + c] - mx);
    se += e;
    dot += e * rv[c];
  }
#pragma unroll
  for (int o = 8; o; o >>= 1) { se += __shfl_xor(se, o, 16); dot += __shfl_xor(dot, o, 16); }
  if (l16 == 0) atomicAdd(&vm_accum[q0 + row], dot / se);
}

// ---------------------------------------------------------------------------
// Out-projection + mish + residual + LayerNorm. One 16-row tile per block.
// ---------------------------------------------------------------------------
__global__ __launch_bounds__(256) void k_outln(const float* __restrict__ X,
                                               const bf16_t* __restrict__ Abf,
                                               const bf16_t* __restrict__ Wt,
                                               const float* __restrict__ bvec,
                                               const float* __restrict__ g,
                                               const float* __restrict__ be,
                                               float* __restrict__ out) {
  __shared__ float sy[16 * 256];
  int lane = threadIdx.x & 31, wave = threadIdx.x >> 5;
  int m0 = blockIdx.x * 16;
  int nloc = lane & 15, half = lane >> 4;
#pragma unroll
  for (int i = 0; i < 2; ++i) {
    int n0 = (wave * 2 + i) * 16;
    v8f acc = {};
#pragma unroll
    for (int kt = 0; kt < 8; ++kt) {
      v16bf a = load_frag_bf16(Abf, D_, m0, kt * 32, lane);
      v16bf b = load_frag_bf16(Wt, D_, n0, kt * 32, lane);
      acc = wmma_bf16(a, b, acc);
    }
    int n = n0 + nloc;
    float bb = bvec[n];
#pragma unroll
    for (int r = 0; r < 8; ++r) {
      int row = r + half * 8;
      sy[row * 256 + n] = X[(long)(m0 + row) * 256 + n] + mish_f(acc[r] + bb);
    }
  }
  __syncthreads();

  int row = threadIdx.x >> 4, l16 = threadIdx.x & 15;
  float s = 0.f;
  for (int c = l16; c < 256; c += 16) s += sy[row * 256 + c];
#pragma unroll
  for (int o = 8; o; o >>= 1) s += __shfl_xor(s, o, 16);
  float mean = s * (1.f / 256.f);
  float vs = 0.f;
  for (int c = l16; c < 256; c += 16) { float d = sy[row * 256 + c] - mean; vs += d * d; }
#pragma unroll
  for (int o = 8; o; o >>= 1) vs += __shfl_xor(vs, o, 16);
  float inv = rsqrtf(vs * (1.f / 256.f) + 1e-5f);
  for (int c = l16; c < 256; c += 16)
    out[(long)(m0 + row) * 256 + c] = (sy[row * 256 + c] - mean) * inv * g[c] + be[c];
}

__global__ void k_finalize(const float* __restrict__ vm, float* __restrict__ out) {
  int i = blockIdx.x * blockDim.x + threadIdx.x;
  if (i < NQ_) out[i] = vm[i] * (1.f / 1024.f);
}

// ---------------------------------------------------------------------------
// Host launcher
// ---------------------------------------------------------------------------
extern "C" void kernel_launch(void* const* d_in, const int* in_sizes, int n_in,
                              void* d_out, int out_size, void* d_ws, size_t ws_size,
                              hipStream_t stream) {
  const float* q     = (const float*)d_in[0];
  const float* k     = (const float*)d_in[1];
  const float* v     = (const float*)d_in[2];
  const float* bias  = (const float*)d_in[3];
  const int*   mask  = (const int*)  d_in[4];
  const float* Wq    = (const float*)d_in[5];
  const float* bq    = (const float*)d_in[6];
  const float* Wk    = (const float*)d_in[7];
  const float* bk    = (const float*)d_in[8];
  const float* Wbias = (const float*)d_in[9];
  const float* bbias = (const float*)d_in[10];
  const float* Wqo   = (const float*)d_in[11];
  const float* bqo   = (const float*)d_in[12];
  const float* Wko   = (const float*)d_in[13];
  const float* bko   = (const float*)d_in[14];
  const float* g_q   = (const float*)d_in[15];
  const float* be_q  = (const float*)d_in[16];
  const float* g_k   = (const float*)d_in[17];
  const float* be_k  = (const float*)d_in[18];
  const float* Wbo   = (const float*)d_in[19];
  const float* bbo   = (const float*)d_in[20];

  char* ws = (char*)d_ws;
  bf16_t* q_bf     = (bf16_t*)(ws + 0);
  bf16_t* k_bf     = (bf16_t*)(ws + 524288);
  bf16_t* projq_bf = (bf16_t*)(ws + 1048576);
  bf16_t* projk_bf = (bf16_t*)(ws + 1572864);
  bf16_t* Wq_t     = (bf16_t*)(ws + 2097152);
  bf16_t* Wk_t     = (bf16_t*)(ws + 2228224);
  bf16_t* Wqo_pt   = (bf16_t*)(ws + 2359296);
  bf16_t* Wko_pt   = (bf16_t*)(ws + 2490368);
  bf16_t* Wbias_t  = (bf16_t*)(ws + 2621440);   // [160 x 64]
  bf16_t* Wbo_t    = (bf16_t*)(ws + 2641920);   // [48 x 160]
  float*  rv       = (float*) (ws + 2657280);
  float*  vm       = (float*) (ws + 2661376);
  float*  diffs    = (float*) (ws + 2665472);   // 16 MB

  float* out      = (float*)d_out;
  float* q_out    = out;                 // 262144
  float* k_out    = out + 262144;        // 262144
  float* vm_out   = out + 524288;        // 1024
  float* bias_out = out + 525312;        // 41943040

  // ---- prep ----
  k_cvt_bf16<<<512, 256, 0, stream>>>(q, q_bf, NQ_ * QDIM_);
  k_cvt_bf16<<<512, 256, 0, stream>>>(k, k_bf, NK_ * QDIM_);
  k_transpose_pad_bf16<<<256, 256, 0, stream>>>(Wq, Wq_t, 256, 256, 256, 256);
  k_transpose_pad_bf16<<<256, 256, 0, stream>>>(Wk, Wk_t, 256, 256, 256, 256);
  k_transpose_pad_bf16<<<64, 256, 0, stream>>>(Wbias, Wbias_t, BD_, HBD_, 64, HBD_);
  k_transpose_pad_bf16<<<64, 256, 0, stream>>>(Wbo, Wbo_t, HBD_, BD_, HBD_, 48);
  k_perm_transpose<<<256, 256, 0, stream>>>(Wqo, Wqo_pt);
  k_perm_transpose<<<256, 256, 0, stream>>>(Wko, Wko_pt);
  k_rowsum<<<4, 256, 0, stream>>>(v, rv);
  hipMemsetAsync(vm, 0, NQ_ * sizeof(float), stream);

  // ---- projections ----
  k_proj<<<128, 256, 0, stream>>>(q_bf, Wq_t, bq, projq_bf);
  k_proj<<<128, 256, 0, stream>>>(k_bf, Wk_t, bk, projk_bf);

  // ---- fused bias path (heavy: 336 MB HBM, async->LDS + bf16 WMMA) ----
  k_bias<<<NPAIR_ / 64, 128, 0, stream>>>(bias, bbias, bbo, Wbias_t, Wbo_t, diffs, bias_out);

  // ---- attention (logits WMMA + softmax + rv dot) ----
  k_attn<<<dim3(64, 4), 256, 0, stream>>>(projq_bf, projk_bf, diffs, mask, rv, vm);

  // ---- out-projection + LayerNorm ----
  k_outln<<<64, 256, 0, stream>>>(q, projq_bf, Wqo_pt, bqo, g_q, be_q, q_out);
  k_outln<<<64, 256, 0, stream>>>(k, projk_bf, Wko_pt, bko, g_k, be_k, k_out);

  // ---- vals_mean ----
  k_finalize<<<4, 256, 0, stream>>>(vm, vm_out);

  (void)in_sizes; (void)n_in; (void)out_size; (void)ws_size;
}
